// MultiHeadAttention_39170101739993
// MI455X (gfx1250) — matrix operand order repaired and checked by reading, not compiled, not run on hardware
//
#include <hip/hip_runtime.h>
#include <hip/hip_bf16.h>

typedef __attribute__((ext_vector_type(16))) _Float16 v16h;
typedef __attribute__((ext_vector_type(8)))  float    v8f;

#define WMMA_F32_F16(A_, B_, C_) \
  __builtin_amdgcn_wmma_f32_16x16x32_f16(false, (A_), false, (B_), (short)0, (C_), false, false)

constexpr int BB = 4;      // batch
constexpr int T  = 2048;   // seq len
constexpr int DD = 512;    // model dim
constexpr int HH = 8;      // heads
constexpr int HD = 64;     // head dim
constexpr float SCALE = 0.044194173824159216f;  // 512^-0.5 (module scales by full d_k)

// A-fragment K-pair start for 16x16x32 f16 WMMA: contiguous k = 16*hh + 2v (B loads match)
__device__ __forceinline__ int a_koff(int v, int hh) {
  return (hh << 4) + (v << 1);
}

// ---------------------------------------------------------------------------
// GEMM 1: X[f32, M x 512] @ W[f32, 512 x 512] + bias -> f16 in per-head
// layout [B][H][T][64].  Block tile 64(M) x 64(N), 8 waves (4x2), each wave a
// 16x32 C tile = 2 WMMAs sharing one A fragment.  K staged in LDS as f16.
// ---------------------------------------------------------------------------
__global__ __launch_bounds__(256) void qkv_proj_kernel(
    const float* __restrict__ X, const float* __restrict__ W,
    const float* __restrict__ bias, _Float16* __restrict__ outPH)
{
  __shared__ __align__(16) _Float16 As[64 * 32];   // [row][k]
  __shared__ __align__(16) _Float16 Bs[64 * 32];   // [n][k]  (transposed: packed K pairs)

  const int tid  = threadIdx.x;
  const int lane = tid & 31;
  const int w    = tid >> 5;
  const int wm   = w >> 1;          // 0..3 (M sub-tile)
  const int wn   = w & 1;           // 0..1 (32-wide N sub-tile)
  const int m0   = blockIdx.x * 64;
  const int n0   = blockIdx.y * 64;
  const int ln   = lane & 15;
  const int hh   = lane >> 4;

  v8f acc[2] = {};

  for (int k0 = 0; k0 < DD; k0 += 32) {
    __syncthreads();
    {   // stage A tile: 64 rows x 32 k (8 f32 -> f16 per thread)
      const int row = tid >> 2;
      const int ks  = (tid & 3) * 8;
      const float* g = X + (size_t)(m0 + row) * DD + k0 + ks;
      __builtin_prefetch(g + 32, 0, 3);            // next K slab (speculative)
      float4 f0 = *(const float4*)g;
      float4 f1 = *(const float4*)(g + 4);
      _Float16* s = As + row * 32 + ks;
      s[0] = (_Float16)f0.x; s[1] = (_Float16)f0.y;
      s[2] = (_Float16)f0.z; s[3] = (_Float16)f0.w;
      s[4] = (_Float16)f1.x; s[5] = (_Float16)f1.y;
      s[6] = (_Float16)f1.z; s[7] = (_Float16)f1.w;
    }
    {   // stage B tile transposed: W[k0+kr][n0+ns..+7] -> Bs[n][k]
      const int kr = tid >> 3;
      const int ns = (tid & 7) * 8;
      const float* gw = W + (size_t)(k0 + kr) * DD + n0 + ns;
      __builtin_prefetch(gw + 32 * DD, 0, 3);      // next K slab (speculative)
      float4 f0 = *(const float4*)gw;
      float4 f1 = *(const float4*)(gw + 4);
      Bs[(ns + 0) * 32 + kr] = (_Float16)f0.x;
      Bs[(ns + 1) * 32 + kr] = (_Float16)f0.y;
      Bs[(ns + 2) * 32 + kr] = (_Float16)f0.z;
      Bs[(ns + 3) * 32 + kr] = (_Float16)f0.w;
      Bs[(ns + 4) * 32 + kr] = (_Float16)f1.x;
      Bs[(ns + 5) * 32 + kr] = (_Float16)f1.y;
      Bs[(ns + 6) * 32 + kr] = (_Float16)f1.z;
      Bs[(ns + 7) * 32 + kr] = (_Float16)f1.w;
    }
    __syncthreads();

    v16h a, b0, b1;
#pragma unroll
    for (int v = 0; v < 8; ++v) {
      const int ka = a_koff(v, hh);
      const _Float16* ap = &As[(wm * 16 + ln) * 32 + ka];
      a[2 * v] = ap[0]; a[2 * v + 1] = ap[1];
      const int kb = (hh << 4) + (v << 1);
      const _Float16* bp0 = &Bs[(wn * 32 + ln) * 32 + kb];
      const _Float16* bp1 = &Bs[(wn * 32 + 16 + ln) * 32 + kb];
      b0[2 * v] = bp0[0]; b0[2 * v + 1] = bp0[1];
      b1[2 * v] = bp1[0]; b1[2 * v + 1] = bp1[1];
    }
    acc[0] = WMMA_F32_F16(a, b0, acc[0]);
    acc[1] = WMMA_F32_F16(a, b1, acc[1]);
  }

#pragma unroll
  for (int nt = 0; nt < 2; ++nt) {
    const int col  = n0 + wn * 32 + nt * 16 + ln;
    const float bv = bias[col];
    const int hidx = col >> 6;      // head
    const int hd   = col & 63;      // dim within head
#pragma unroll
    for (int v = 0; v < 8; ++v) {
      const int row = m0 + wm * 16 + v + 8 * hh;   // C-frag row = v + 8*half
      const int bb  = row >> 11;                   // row / T
      const int t   = row & (T - 1);
      outPH[(((size_t)bb * HH + hidx) * T + t) * HD + hd] = (_Float16)(acc[nt][v] + bv);
    }
  }
}

// ---------------------------------------------------------------------------
// GEMM 2: Aattn[f16, M x 512] @ Wo[f32] + bo -> f32 row-major output.
// Same 64x64 block / 16x32 wave tiling.
// ---------------------------------------------------------------------------
__global__ __launch_bounds__(256) void out_proj_kernel(
    const _Float16* __restrict__ Ah, const float* __restrict__ W,
    const float* __restrict__ bias, float* __restrict__ out)
{
  __shared__ __align__(16) _Float16 As[64 * 32];
  __shared__ __align__(16) _Float16 Bs[64 * 32];

  const int tid  = threadIdx.x;
  const int lane = tid & 31;
  const int w    = tid >> 5;
  const int wm   = w >> 1;
  const int wn   = w & 1;
  const int m0   = blockIdx.x * 64;
  const int n0   = blockIdx.y * 64;
  const int ln   = lane & 15;
  const int hh   = lane >> 4;

  v8f acc[2] = {};

  for (int k0 = 0; k0 < DD; k0 += 32) {
    __syncthreads();
    {   // stage A (already f16): 16-byte vector copy
      const int row = tid >> 2;
      const int ks  = (tid & 3) * 8;
      const _Float16* g = Ah + (size_t)(m0 + row) * DD + k0 + ks;
      __builtin_prefetch(g + 32, 0, 3);
      uint4 u = *(const uint4*)g;
      *(uint4*)(As + row * 32 + ks) = u;
    }
    {   // stage B transposed
      const int kr = tid >> 3;
      const int ns = (tid & 7) * 8;
      const float* gw = W + (size_t)(k0 + kr) * DD + n0 + ns;
      __builtin_prefetch(gw + 32 * DD, 0, 3);
      float4 f0 = *(const float4*)gw;
      float4 f1 = *(const float4*)(gw + 4);
      Bs[(ns + 0) * 32 + kr] = (_Float16)f0.x;
      Bs[(ns + 1) * 32 + kr] = (_Float16)f0.y;
      Bs[(ns + 2) * 32 + kr] = (_Float16)f0.z;
      Bs[(ns + 3) * 32 + kr] = (_Float16)f0.w;
      Bs[(ns + 4) * 32 + kr] = (_Float16)f1.x;
      Bs[(ns + 5) * 32 + kr] = (_Float16)f1.y;
      Bs[(ns + 6) * 32 + kr] = (_Float16)f1.z;
      Bs[(ns + 7) * 32 + kr] = (_Float16)f1.w;
    }
    __syncthreads();

    v16h a, b0, b1;
#pragma unroll
    for (int v = 0; v < 8; ++v) {
      const int ka = a_koff(v, hh);
      const _Float16* ap = &As[(wm * 16 + ln) * 32 + ka];
      a[2 * v] = ap[0]; a[2 * v + 1] = ap[1];
      const int kb = (hh << 4) + (v << 1);
      const _Float16* bp0 = &Bs[(wn * 32 + ln) * 32 + kb];
      const _Float16* bp1 = &Bs[(wn * 32 + 16 + ln) * 32 + kb];
      b0[2 * v] = bp0[0]; b0[2 * v + 1] = bp0[1];
      b1[2 * v] = bp1[0]; b1[2 * v + 1] = bp1[1];
    }
    acc[0] = WMMA_F32_F16(a, b0, acc[0]);
    acc[1] = WMMA_F32_F16(a, b1, acc[1]);
  }

#pragma unroll
  for (int nt = 0; nt < 2; ++nt) {
    const int col  = n0 + wn * 32 + nt * 16 + ln;
    const float bv = bias[col];
#pragma unroll
    for (int v = 0; v < 8; ++v) {
      const int row = m0 + wm * 16 + v + 8 * hh;
      out[(size_t)row * DD + col] = acc[nt][v] + bv;
    }
  }
}

// ---------------------------------------------------------------------------
// Flash-style causal attention.  Grid: (T/128, B*H).  8 waves/block, one
// 16-row q-tile per wave.  64-key chunks of K/V staged cooperatively in LDS
// (16 WMMAs per barrier pair).  Q/K/V f16 in [B][H][T][64]; out f16 [B][T][512].
// ---------------------------------------------------------------------------
__global__ __launch_bounds__(256) void attn_kernel(
    const _Float16* __restrict__ Qh, const _Float16* __restrict__ Kh,
    const _Float16* __restrict__ Vh, _Float16* __restrict__ attnOut)
{
  __shared__ __align__(16) _Float16 Ks[64 * 64];     // [key][dim]       8 KB
  __shared__ __align__(16) _Float16 Vt[64 * 64];     // [dim][key]       8 KB
  __shared__ __align__(16) _Float16 Ps[8][16 * 64];  // per-wave P      16 KB

  const int tid  = threadIdx.x;
  const int lane = tid & 31;
  // wave id made explicitly uniform so the causal skip is a scalar branch
  // (WMMA requires EXEC all-ones; never run WMMA under a lane-masked branch)
  const int w    = __builtin_amdgcn_readfirstlane(tid >> 5);
  const int ln   = lane & 15;
  const int hh   = lane >> 4;
  const int bh   = blockIdx.y;
  const int b    = bh >> 3;
  const int h    = bh & 7;
  const int qt   = blockIdx.x * 8 + w;              // q-tile index (16 rows)
  const size_t base = (size_t)bh * T * HD;

  // Preload Q as two A-fragments (head dim 64 = 2 K-chunks of 32)
  v16h Aq[2];
  {
    const _Float16* qrowp = Qh + base + (size_t)(qt * 16 + ln) * HD;
#pragma unroll
    for (int kc = 0; kc < 2; ++kc)
#pragma unroll
      for (int v = 0; v < 8; ++v) {
        const int ka = kc * 32 + a_koff(v, hh);
        Aq[kc][2 * v]     = qrowp[ka];
        Aq[kc][2 * v + 1] = qrowp[ka + 1];
      }
  }

  float m[8], l[8];
  v8f o[4] = {};
#pragma unroll
  for (int v = 0; v < 8; ++v) { m[v] = -3.0e38f; l[v] = 0.0f; }

  const int nch  = (blockIdx.x + 1) * 2;   // 64-key chunks up to block's q range
  const int qmax = qt * 16 + 15;

  for (int c = 0; c < nch; ++c) {
    const int kb = c * 64;
    __syncthreads();
    {   // cooperative staging: 64 keys x 64 dims; K row-major, V transposed
      const int key = tid >> 2;
      const int ds  = (tid & 3) * 16;
      const _Float16* kp = Kh + base + (size_t)(kb + key) * HD + ds;
      uint4 k0 = *(const uint4*)kp;
      uint4 k1 = *(const uint4*)(kp + 8);
      *(uint4*)(Ks + key * 64 + ds)     = k0;
      *(uint4*)(Ks + key * 64 + ds + 8) = k1;
      const _Float16* vp = Vh + base + (size_t)(kb + key) * HD + ds;
      union { uint4 u[2]; _Float16 e[16]; } vbuf;
      vbuf.u[0] = *(const uint4*)vp;
      vbuf.u[1] = *(const uint4*)(vp + 8);
#pragma unroll
      for (int j = 0; j < 16; ++j) Vt[(ds + j) * 64 + key] = vbuf.e[j];
    }
    __syncthreads();
    if (kb > qmax) continue;   // wave-uniform: chunk fully above the diagonal

    // S = Q . K^T over 64 keys (four 16-wide key tiles, 8 WMMAs)
    v8f s[4] = {};
#pragma unroll
    for (int kc = 0; kc < 2; ++kc) {
#pragma unroll
      for (int nt = 0; nt < 4; ++nt) {
        v16h bf;
#pragma unroll
        for (int v = 0; v < 8; ++v) {
          const int kd = kc * 32 + (hh << 4) + (v << 1);
          const _Float16* p = &Ks[(nt * 16 + ln) * 64 + kd];
          bf[2 * v] = p[0]; bf[2 * v + 1] = p[1];
        }
        s[nt] = WMMA_F32_F16(Aq[kc], bf, s[nt]);
      }
    }

    // scale + causal mask + online softmax (row lives in 16 lanes -> xor 1..8)
    float alpha[8];
#pragma unroll
    for (int v = 0; v < 8; ++v) {
      const int qrow = qt * 16 + v + 8 * hh;
      float x[4];
#pragma unroll
      for (int nt = 0; nt < 4; ++nt)
        x[nt] = s[nt][v] * SCALE + ((kb + nt * 16 + ln) > qrow ? -1.0e9f : 0.0f);
      float cm = fmaxf(fmaxf(x[0], x[1]), fmaxf(x[2], x[3]));
      cm = fmaxf(cm, __shfl_xor(cm, 1, 32));
      cm = fmaxf(cm, __shfl_xor(cm, 2, 32));
      cm = fmaxf(cm, __shfl_xor(cm, 4, 32));
      cm = fmaxf(cm, __shfl_xor(cm, 8, 32));
      const float mn = fmaxf(m[v], cm);
      const float al = expf(m[v] - mn);
      float psum = 0.0f;
#pragma unroll
      for (int nt = 0; nt < 4; ++nt) {
        const float p = expf(x[nt] - mn);
        psum += p;
        Ps[w][(v + 8 * hh) * 64 + nt * 16 + ln] = (_Float16)p;  // C-layout -> LDS
      }
      l[v] = l[v] * al + psum;
      m[v] = mn;
      alpha[v] = al;
    }
#pragma unroll
    for (int nt = 0; nt < 4; ++nt)
#pragma unroll
      for (int v = 0; v < 8; ++v) o[nt][v] *= alpha[v];

    // P re-read as two A-fragments (wave-internal LDS round trip)
    v16h pa[2];
#pragma unroll
    for (int kc = 0; kc < 2; ++kc)
#pragma unroll
      for (int v = 0; v < 8; ++v) {
        const int ka = kc * 32 + a_koff(v, hh);
        const _Float16* pp = &Ps[w][ln * 64 + ka];
        pa[kc][2 * v]     = pp[0];
        pa[kc][2 * v + 1] = pp[1];
      }
    // O += P . V  (4 head-dim N-tiles x 2 key K-chunks = 8 WMMAs)
#pragma unroll
    for (int nt = 0; nt < 4; ++nt) {
#pragma unroll
      for (int kc = 0; kc < 2; ++kc) {
        v16h vb;
#pragma unroll
        for (int v = 0; v < 8; ++v) {
          const _Float16* vp = &Vt[(nt * 16 + ln) * 64 + kc * 32 + (hh << 4) + (v << 1)];
          vb[2 * v] = vp[0]; vb[2 * v + 1] = vp[1];
        }
        o[nt] = WMMA_F32_F16(pa[kc], vb, o[nt]);
      }
    }
  }

  // finalize: reduce l across the 16 lanes of each row, normalize, store
#pragma unroll
  for (int v = 0; v < 8; ++v) {
    float lt = l[v];
    lt += __shfl_xor(lt, 1, 32);
    lt += __shfl_xor(lt, 2, 32);
    lt += __shfl_xor(lt, 4, 32);
    lt += __shfl_xor(lt, 8, 32);
    l[v] = 1.0f / lt;
  }
#pragma unroll
  for (int nt = 0; nt < 4; ++nt)
#pragma unroll
    for (int v = 0; v < 8; ++v) {
      const int row = qt * 16 + v + 8 * hh;
      const int col = h * HD + nt * 16 + ln;
      attnOut[((size_t)b * T + row) * DD + col] = (_Float16)(o[nt][v] * l[v]);
    }
}

// ---------------------------------------------------------------------------
extern "C" void kernel_launch(void* const* d_in, const int* in_sizes, int n_in,
                              void* d_out, int out_size, void* d_ws, size_t ws_size,
                              hipStream_t stream) {
  (void)in_sizes; (void)n_in; (void)out_size; (void)ws_size;

  const float* q    = (const float*)d_in[0];
  const float* k    = (const float*)d_in[1];
  const float* vv   = (const float*)d_in[2];
  // d_in[3] is the causal mask; causality is applied analytically in-kernel.
  const float* Wq   = (const float*)d_in[4];
  const float* bq   = (const float*)d_in[5];
  const float* Wk   = (const float*)d_in[6];
  const float* bk   = (const float*)d_in[7];
  const float* Wv   = (const float*)d_in[8];
  const float* bv   = (const float*)d_in[9];
  const float* Wo   = (const float*)d_in[10];
  const float* bo   = (const float*)d_in[11];
  float* out        = (float*)d_out;

  const size_t nQKV = (size_t)BB * HH * T * HD;   // 4,194,304 f16 elements each
  _Float16* Qh   = (_Float16*)d_ws;
  _Float16* Kh   = Qh + nQKV;
  _Float16* Vh   = Kh + nQKV;
  _Float16* attn = Vh + nQKV;                      // [B][T][512] f16

  const dim3 gblk(256);
  const dim3 ggrid((BB * T) / 64, DD / 64);        // 128 x 8

  qkv_proj_kernel<<<ggrid, gblk, 0, stream>>>(q,  Wq, bq, Qh);
  qkv_proj_kernel<<<ggrid, gblk, 0, stream>>>(k,  Wk, bk, Kh);
  qkv_proj_kernel<<<ggrid, gblk, 0, stream>>>(vv, Wv, bv, Vh);

  const dim3 agrid(T / 128, BB * HH);              // 16 x 32
  attn_kernel<<<agrid, gblk, 0, stream>>>(Qh, Kh, Vh, attn);

  out_proj_kernel<<<ggrid, gblk, 0, stream>>>(attn, Wo, bo, out);
}
